// DecoderVCSC_69372311765126
// MI455X (gfx1250) — matrix operand order repaired and checked by reading, not compiled, not run on hardware
//
#include <hip/hip_runtime.h>

#define G_    1024
#define NIN_  64
#define H_    128
#define B_    256
#define P_    523776            // G*(G-1)/2
#define KBLK  1536              // pairs per workgroup K-slice; P_/KBLK = 341
#define NSLICE (P_ / KBLK)      // 341
#define BN_EPS 1e-3f

typedef __attribute__((ext_vector_type(16))) __bf16 v16bf;
typedef __attribute__((ext_vector_type(8)))  __bf16 v8bf;
typedef __attribute__((ext_vector_type(8)))  float  v8f;

// ---------------------------------------------------------------- pair indices
__global__ void pair_idx_kernel(int* __restrict__ iu, int* __restrict__ jv) {
    const int p = blockIdx.x * blockDim.x + threadIdx.x;
    if (p >= P_) return;
    // largest i with off(i) = i*(2G-i-1)/2 <= p
    const double t = 2.0 * G_ - 1.0;
    const double disc = t * t - 8.0 * (double)p;
    int i = (int)((t - sqrt(disc)) * 0.5);
    if (i < 0) i = 0;
    if (i > G_ - 2) i = G_ - 2;
    while (i > 0 && (long long)i * (2 * G_ - i - 1) / 2 > (long long)p) --i;
    while (i < G_ - 2 && (long long)(i + 1) * (2 * G_ - i - 2) / 2 <= (long long)p) ++i;
    const long long off = (long long)i * (2 * G_ - i - 1) / 2;
    iu[p] = i;
    jv[p] = (int)((long long)p - off) + i + 1;
}

// ---------------------------------------------------------------- c^T = (z @ Wc.T + bc)^T, stored G x B
__global__ void cT_kernel(const float* __restrict__ z, const float* __restrict__ Wc,
                          const float* __restrict__ bc, float* __restrict__ cT) {
    const int g = blockIdx.x;      // 0..G-1
    const int b = threadIdx.x;     // 0..B-1
    const float* zr = z + b * NIN_;
    const float* wr = Wc + g * NIN_;
    float acc = 0.f;
#pragma unroll
    for (int k = 0; k < NIN_; ++k) acc = fmaf(zr[k], wr[k], acc);
    cT[g * B_ + b] = acc + bc[g];
}

__global__ void zero_kernel(float* __restrict__ p, int n) {
    const int t = blockIdx.x * blockDim.x + threadIdx.x;
    if (t < n) p[t] = 0.f;
}

// ---------------------------------------------------------------- fused SDP + split-K GEMM
// Each workgroup: one K-slice (1536 pairs) x full 256x128 output tile.
// Software-pipelined: double-buffered LDS, W1 prefetched into registers
// across the WMMA block, one barrier per k-step. px_sigma written once
// (nontemporal), W1 read once (nontemporal).
__launch_bounds__(256, 1)
__global__ void sdp_gemm_kernel(const float* __restrict__ cT,
                                const int*   __restrict__ iu,
                                const int*   __restrict__ jv,
                                const float* __restrict__ W1,     // H_ x P_
                                float*       __restrict__ sigma,  // B_ x P_ (output)
                                float*       __restrict__ hacc)   // B_ x H_ (atomic accum)
{
    __shared__ __align__(16) __bf16 sA[2][B_ * 40];   // batch rows, stride 40 elem = 80 B
    __shared__ __align__(16) __bf16 sB[2][H_ * 40];   // H rows

    const int tid  = threadIdx.x;
    const int lane = tid & 31;
    const int wave = tid >> 5;                 // 0..7, owns batch rows [32w, 32w+32)
    const int k0   = blockIdx.x * KBLK;

    v8f acc[2][8] = {};                        // 2 M-tiles x 8 N-tiles of 16x16 f32

    const int kA     = tid & 31;               // k this thread stages for A
    const int nB     = tid >> 1;               // H row this thread stages for B
    const int kBoff  = (tid & 1) * 16;

    const int rsel  = lane & 15;
    const int half  = lane >> 4;               // 0: lanes 0-15, 1: lanes 16-31
    const int c0    = half * 8;                // unused; A also reads K {kbase..kbase+15}
    const int kbase = half * 16;               // B chunk base (K {kbase..kbase+15})

    float wpre[16];
    int   ipre, jpre;

    // --- load phase: issue next k-step's global traffic (W1 + pair indices)
    auto loadPhase = [&](int kp) {
        const float* src = W1 + (size_t)nB * P_ + kp + kBoff;
#pragma unroll
        for (int q = 0; q < 16; ++q) wpre[q] = __builtin_nontemporal_load(src + q);
        ipre = iu[kp + kA];
        jpre = jv[kp + kA];
    };
    // --- store phases: convert & populate LDS buffers, emit px_sigma
    auto storeB = [&](int buf) {
        __bf16* dst = &sB[buf][nB * 40 + kBoff];
#pragma unroll
        for (int q = 0; q < 16; ++q) dst[q] = (__bf16)wpre[q];
    };
    auto stageA = [&](int buf, int kp) {
        const int p = kp + kA;
        const float* ci = cT + (size_t)ipre * B_;
        const float* cj = cT + (size_t)jpre * B_;
#pragma unroll
        for (int it = 0; it < 32; ++it) {
            const int b = it * 8 + wave;
            const float s = __expf(cj[b] - ci[b]);   // exp(-(c_i - c_j)), ALPHA=1
            __builtin_nontemporal_store(s, &sigma[(size_t)b * P_ + p]);
            sA[buf][b * 40 + kA] = (__bf16)s;
        }
    };

    // prologue: fully stage buffer 0
    loadPhase(k0);
    storeB(0);
    stageA(0, k0);

    int cur = 0;
    for (int kb = 0; kb < KBLK; kb += 32) {
        __syncthreads();   // buf[cur] staged; buf[cur^1] free for restage
        const bool hasNext = (kb + 32) < KBLK;
        if (hasNext) loadPhase(k0 + kb + 32);          // global loads in flight over WMMAs

        // ---- WMMA on buf[cur]
        v16bf afrag[2];
#pragma unroll
        for (int mt = 0; mt < 2; ++mt) {
            const __bf16* rowA = &sA[cur][(wave * 32 + mt * 16 + rsel) * 40];
            v8bf lo = *(const v8bf*)(rowA + kbase);
            v8bf hi = *(const v8bf*)(rowA + kbase + 8);
            afrag[mt] = __builtin_shufflevector(lo, hi,
                0, 1, 2, 3, 4, 5, 6, 7, 8, 9, 10, 11, 12, 13, 14, 15);
        }
#pragma unroll
        for (int nt = 0; nt < 8; ++nt) {
            const __bf16* rowB = &sB[cur][(nt * 16 + rsel) * 40 + kbase];
            v8bf blo = *(const v8bf*)(rowB);
            v8bf bhi = *(const v8bf*)(rowB + 8);
            v16bf bfrag = __builtin_shufflevector(blo, bhi,
                0, 1, 2, 3, 4, 5, 6, 7, 8, 9, 10, 11, 12, 13, 14, 15);
#pragma unroll
            for (int mt = 0; mt < 2; ++mt)
                acc[mt][nt] = __builtin_amdgcn_wmma_f32_16x16x32_bf16(
                    false, afrag[mt], false, bfrag, (short)0, acc[mt][nt], false, false);
        }

        // ---- restage alternate buffer for next k-step
        if (hasNext) {
            storeB(cur ^ 1);
            stageA(cur ^ 1, k0 + kb + 32);
        }
        cur ^= 1;
    }

    // ---- split-K reduction: atomic-add partial C into hacc
#pragma unroll
    for (int mt = 0; mt < 2; ++mt)
#pragma unroll
        for (int nt = 0; nt < 8; ++nt)
#pragma unroll
            for (int r = 0; r < 8; ++r) {
                const int m = wave * 32 + mt * 16 + half * 8 + r;   // C layout: VGPR r -> M=r (+8 hi half)
                const int n = nt * 16 + rsel;
                unsafeAtomicAdd(&hacc[m * H_ + n], acc[mt][nt][r]);
            }
}

// ---------------------------------------------------------------- BatchNorm (training stats) + ReLU
__global__ void bn_relu_kernel(const float* __restrict__ hacc, const float* __restrict__ b1,
                               const float* __restrict__ gamma, const float* __restrict__ beta,
                               float* __restrict__ hact) {
    const int t = threadIdx.x;   // column 0..H_-1
    const float bias = b1[t];
    float s = 0.f, s2 = 0.f;
    for (int b = 0; b < B_; ++b) {
        const float x = hacc[b * H_ + t] + bias;
        s += x; s2 += x * x;
    }
    const float mu  = s * (1.f / B_);
    const float var = s2 * (1.f / B_) - mu * mu;
    const float inv = rsqrtf(var + BN_EPS);
    const float ga = gamma[t], be = beta[t];
    for (int b = 0; b < B_; ++b) {
        const float x = hacc[b * H_ + t] + bias;
        const float y = ga * (x - mu) * inv + be;
        hact[b * H_ + t] = y > 0.f ? y : 0.f;
    }
}

// ---------------------------------------------------------------- heads: h @ W.T + b (shape / scale-logits / dropout)
__global__ void heads_kernel(const float* __restrict__ hact,
                             const float* __restrict__ Wshape, const float* __restrict__ bshape,
                             const float* __restrict__ Wscale, const float* __restrict__ bscale,
                             const float* __restrict__ Wdrop,  const float* __restrict__ bdrop,
                             float* __restrict__ out) {
    __shared__ float hrow[H_];
    const int b = blockIdx.x;
    const int head = blockIdx.y;      // 0: shape, 1: scale logits, 2: dropout
    const int t = threadIdx.x;        // 0..255
    if (t < H_) hrow[t] = hact[b * H_ + t];
    __syncthreads();
    const float* W  = head == 0 ? Wshape : head == 1 ? Wscale : Wdrop;
    const float* bb = head == 0 ? bshape : head == 1 ? bscale : bdrop;
    float* dst = out + (size_t)head * (B_ * G_) + (size_t)b * G_;
#pragma unroll
    for (int q = 0; q < G_ / 256; ++q) {
        const int g = q * 256 + t;
        const float* wr = W + (size_t)g * H_;
        float acc = bb[g];
#pragma unroll
        for (int k = 0; k < H_; ++k) acc = fmaf(hrow[k], wr[k], acc);
        dst[g] = acc;
    }
}

// ---------------------------------------------------------------- row softmax (in place) for px_scale
__global__ void softmax_kernel(float* __restrict__ scale) {
    __shared__ float red[256];
    const int b = blockIdx.x, t = threadIdx.x;
    float* row = scale + (size_t)b * G_;
    float v[4];
    float m = -INFINITY;
#pragma unroll
    for (int q = 0; q < 4; ++q) { v[q] = row[q * 256 + t]; m = fmaxf(m, v[q]); }
    red[t] = m; __syncthreads();
    for (int s = 128; s > 0; s >>= 1) {
        if (t < s) red[t] = fmaxf(red[t], red[t + s]);
        __syncthreads();
    }
    m = red[0]; __syncthreads();
    float sum = 0.f;
#pragma unroll
    for (int q = 0; q < 4; ++q) { v[q] = __expf(v[q] - m); sum += v[q]; }
    red[t] = sum; __syncthreads();
    for (int s = 128; s > 0; s >>= 1) {
        if (t < s) red[t] += red[t + s];
        __syncthreads();
    }
    const float inv = 1.f / red[0];
#pragma unroll
    for (int q = 0; q < 4; ++q) row[q * 256 + t] = v[q] * inv;
}

// ---------------------------------------------------------------- launch
extern "C" void kernel_launch(void* const* d_in, const int* in_sizes, int n_in,
                              void* d_out, int out_size, void* d_ws, size_t ws_size,
                              hipStream_t stream) {
    (void)in_sizes; (void)n_in; (void)out_size; (void)ws_size;
    const float* z      = (const float*)d_in[0];
    const float* Wc     = (const float*)d_in[1];
    const float* bc     = (const float*)d_in[2];
    const float* W1     = (const float*)d_in[3];
    const float* b1     = (const float*)d_in[4];
    const float* gamma  = (const float*)d_in[5];
    const float* beta   = (const float*)d_in[6];
    const float* Wscale = (const float*)d_in[7];
    const float* bscale = (const float*)d_in[8];
    const float* Wshape = (const float*)d_in[9];
    const float* bshape = (const float*)d_in[10];
    const float* Wdrop  = (const float*)d_in[11];
    const float* bdrop  = (const float*)d_in[12];

    float* out       = (float*)d_out;
    float* out_scale = out + (size_t)B_ * G_;        // tuple order: shape, scale, dropout, sigma
    float* out_sigma = out + (size_t)3 * B_ * G_;

    char* ws = (char*)d_ws;
    int*   iu   = (int*)ws;    ws += sizeof(int)   * (size_t)P_;
    int*   jv   = (int*)ws;    ws += sizeof(int)   * (size_t)P_;
    float* cT   = (float*)ws;  ws += sizeof(float) * (size_t)G_ * B_;
    float* hacc = (float*)ws;  ws += sizeof(float) * (size_t)B_ * H_;
    float* hact = (float*)ws;

    hipLaunchKernelGGL(pair_idx_kernel, dim3(P_ / 256), dim3(256), 0, stream, iu, jv);
    hipLaunchKernelGGL(cT_kernel,       dim3(G_),       dim3(B_),  0, stream, z, Wc, bc, cT);
    hipLaunchKernelGGL(zero_kernel,     dim3((B_ * H_) / 256), dim3(256), 0, stream, hacc, B_ * H_);
    hipLaunchKernelGGL(sdp_gemm_kernel, dim3(NSLICE),   dim3(256), 0, stream,
                       cT, iu, jv, W1, out_sigma, hacc);
    hipLaunchKernelGGL(bn_relu_kernel,  dim3(1),        dim3(H_),  0, stream,
                       hacc, b1, gamma, beta, hact);
    hipLaunchKernelGGL(heads_kernel,    dim3(B_, 3),    dim3(256), 0, stream,
                       hact, Wshape, bshape, Wscale, bscale, Wdrop, bdrop, out);
    hipLaunchKernelGGL(softmax_kernel,  dim3(B_),       dim3(256), 0, stream, out_scale);
}
